// MLAAttention_40802189312606
// MI455X (gfx1250) — matrix operand order repaired and checked by reading, not compiled, not run on hardware
//
#include <hip/hip_runtime.h>
#include <hip/hip_bf16.h>
#include <stdint.h>

// ---------------------------------------------------------------------------
// MLA attention forward for gfx1250 (MI455X), f16 WMMA with f32 accumulation.
// All WMMA fragments are contiguous 16/32-byte runs (weights & V transposed).
// ---------------------------------------------------------------------------

typedef __attribute__((ext_vector_type(16))) _Float16 v16h;
typedef __attribute__((ext_vector_type(8)))  _Float16 v8h;
typedef __attribute__((ext_vector_type(8)))  float    v8f;

#define B_DIM 2
#define L_DIM 2048
#define D_DIM 2048
#define H_DIM 16
#define DC_DIM 512
#define DH_DIM 128
#define ROWS (B_DIM * L_DIM)   // 4096
#define ATTN_SCALE 0.0625f     // 1/sqrt(2*DH) = 1/16

__device__ __forceinline__ v8f wmma_f16(v16h a, v16h b, v8f c) {
  return __builtin_amdgcn_wmma_f32_16x16x32_f16(
      /*neg_a=*/false, a, /*neg_b=*/false, b,
      /*c_mod=*/(short)0, c, /*reuse_a=*/false, /*reuse_b=*/false);
}

__device__ __forceinline__ v16h join8(v8h lo, v8h hi) {
  v16h r;
#pragma unroll
  for (int i = 0; i < 8; ++i) { r[i] = lo[i]; r[8 + i] = hi[i]; }
  return r;
}

// ---------------------------------------------------------------------------
// f32 -> f16 convert (plain)
// ---------------------------------------------------------------------------
__global__ void cvt_f32_to_f16(const float* __restrict__ in,
                               _Float16* __restrict__ out, int n) {
  int i = blockIdx.x * blockDim.x + threadIdx.x;
  int stride = gridDim.x * blockDim.x;
  for (; i < n; i += stride) out[i] = (_Float16)in[i];
}

// f32 [K,N] -> f16 transposed [N,K], LDS-tiled so both sides are coalesced.
// Block 32x8, tile 32x32. K, N multiples of 32.
__global__ void cvt_transpose_f16(const float* __restrict__ in,
                                  _Float16* __restrict__ out, int K, int N) {
  __shared__ float tile[32][33];
  const int tx = threadIdx.x, ty = threadIdx.y;
  const int kb = blockIdx.x * 32, nb = blockIdx.y * 32;
#pragma unroll
  for (int i = 0; i < 4; ++i)
    tile[ty + i * 8][tx] = in[(size_t)(kb + ty + i * 8) * N + nb + tx];
  __syncthreads();
#pragma unroll
  for (int i = 0; i < 4; ++i)
    out[(size_t)(nb + ty + i * 8) * K + kb + tx] =
        (_Float16)tile[tx][ty + i * 8];
}

// v_c [B*L, H*DH] f16 -> vt [(b*H+h)*DH + d][L] f16, LDS-tiled.
// Grid: (L/32, DH/32, B*H), block 32x8.
__global__ void transpose_v_f16(const _Float16* __restrict__ vc,
                                _Float16* __restrict__ vt) {
  __shared__ _Float16 tile[32][33];
  const int tx = threadIdx.x, ty = threadIdx.y;
  const int bh = blockIdx.z;
  const int b = bh >> 4, h = bh & (H_DIM - 1);
  const int keyb = blockIdx.x * 32, db = blockIdx.y * 32;
#pragma unroll
  for (int i = 0; i < 4; ++i)
    tile[ty + i * 8][tx] =
        vc[(size_t)(b * L_DIM + keyb + ty + i * 8) * D_DIM + h * DH_DIM + db + tx];
  __syncthreads();
#pragma unroll
  for (int i = 0; i < 4; ++i)
    vt[(size_t)(bh * DH_DIM + db + ty + i * 8) * L_DIM + keyb + tx] =
        tile[tx][ty + i * 8];
}

// ---------------------------------------------------------------------------
// WMMA GEMM:  C[M,N] = A[M,K] (row-major f16) * Bt[N,K] (f16, B transposed)
// Block = 256 threads = 8 waves; block tile 64(M) x 128(N); K step 32.
// A tile (64x32) staged in LDS; each wave: 4 M-subtiles x one 16-col N strip,
// so each 32B B-fragment load feeds 4 WMMAs.
// ---------------------------------------------------------------------------
template <bool HALF_OUT>
__global__ __launch_bounds__(256, 1) void gemm_wmma_f16t(
    const _Float16* __restrict__ A, const _Float16* __restrict__ Bt,
    void* __restrict__ Cv, int M, int N, int K) {
  __shared__ _Float16 As[64 * 32];

  const int tid  = threadIdx.x;
  const int wave = tid >> 5;
  const int lane = tid & 31;
  const int hi   = lane >> 4;
  const int ln   = lane & 15;

  const int m0 = blockIdx.y * 64;
  const int n0 = blockIdx.x * 128 + wave * 16;

  v8f acc[4];
#pragma unroll
  for (int mt = 0; mt < 4; ++mt) acc[mt] = (v8f){};

  const int ar = tid >> 2;        // 0..63  (row of A tile)
  const int ac = (tid & 3) * 8;   // 0,8,16,24 (8-half segment)

  const _Float16* brow = &Bt[(size_t)(n0 + ln) * K + hi * 16];

  for (int kc = 0; kc < K; kc += 32) {
    *reinterpret_cast<v8h*>(&As[ar * 32 + ac]) =
        *reinterpret_cast<const v8h*>(&A[(size_t)(m0 + ar) * K + kc + ac]);
    __syncthreads();

    v16h bf = *reinterpret_cast<const v16h*>(&brow[kc]);
    if (kc + 32 < K) __builtin_prefetch(&brow[kc + 32], 0, 0);

#pragma unroll
    for (int mt = 0; mt < 4; ++mt) {
      v8h alo = *reinterpret_cast<const v8h*>(&As[(mt * 16 + ln) * 32 + hi * 16]);
      v8h ahi = *reinterpret_cast<const v8h*>(&As[(mt * 16 + ln) * 32 + hi * 16 + 8]);
      acc[mt] = wmma_f16(join8(alo, ahi), bf, acc[mt]);
    }
    __syncthreads();
  }

#pragma unroll
  for (int mt = 0; mt < 4; ++mt) {
#pragma unroll
    for (int j = 0; j < 8; ++j) {
      size_t idx = (size_t)(m0 + mt * 16 + j + hi * 8) * N + n0 + ln;
      if (HALF_OUT)
        reinterpret_cast<_Float16*>(Cv)[idx] = (_Float16)acc[mt][j];
      else
        reinterpret_cast<float*>(Cv)[idx] = acc[mt][j];
    }
  }
}

// ---------------------------------------------------------------------------
// RoPE (reference's interleaved indexing reproduced exactly)
// ---------------------------------------------------------------------------
__global__ void rope_apply(const _Float16* __restrict__ pre,
                           _Float16* __restrict__ out) {
  const float LOG2_BASE = 13.28771237954945f;  // log2(10000)
  int t = blockIdx.x * blockDim.x + threadIdx.x;
  if (t >= ROWS * 64) return;
  int row = t >> 6;
  int j   = t & 63;
  float l = (float)(row & (L_DIM - 1));

  int ia = 2 * j, ib = 2 * j + 1;
  float s, c;
  if (ia < 64) s = sinf(l * exp2f(-(2.0f * ia / 128.0f) * LOG2_BASE));
  else         s = cosf(l * exp2f(-(2.0f * (ia - 64) / 128.0f) * LOG2_BASE));
  if (ib < 64) c = sinf(l * exp2f(-(2.0f * ib / 128.0f) * LOG2_BASE));
  else         c = cosf(l * exp2f(-(2.0f * (ib - 64) / 128.0f) * LOG2_BASE));

  float xe = (float)pre[(size_t)row * 128 + 2 * j];
  float xo = (float)pre[(size_t)row * 128 + 2 * j + 1];
  out[(size_t)row * 128 + j]      = (_Float16)(xe * c - xo * s);
  out[(size_t)row * 128 + 64 + j] = (_Float16)(xe * s + xo * c);
}

// ---------------------------------------------------------------------------
// Fused flash attention.
// qc/kc: [ROWS, H*DH], qr/kr: [ROWS, DH] (broadcast over heads)
// vt: [(b*H+h)*DH + d][L]  (V transposed so key is the fast axis)
// Block = 128 threads = 4 waves, each wave owns 16 query rows.
// All LDS regions are per-wave; LDS ops are in-order per wave -> no barriers.
// ---------------------------------------------------------------------------
__global__ __launch_bounds__(128, 1) void mla_flash_attn(
    const _Float16* __restrict__ qc, const _Float16* __restrict__ qr,
    const _Float16* __restrict__ kcm, const _Float16* __restrict__ krm,
    const _Float16* __restrict__ vt, const int* __restrict__ mask,
    _Float16* __restrict__ ctx) {
  __shared__ _Float16 Qt[4][16 * 256];  // per-wave Q tile (16 x 256) = 8KB
  __shared__ _Float16 Pl[4][16 * 32];   // per-wave P tile (16 x 32)  = 1KB

  const int tid  = threadIdx.x;
  const int wave = tid >> 5;
  const int lane = tid & 31;
  const int hi   = lane >> 4;
  const int ln   = lane & 15;

  const int b  = blockIdx.z;
  const int h  = blockIdx.y;
  const int q0 = blockIdx.x * 64 + wave * 16;

  // ---- stage Q tile (concat q_c | q_r) into per-wave LDS, row-major [16,256]
  {
    const int qrow = b * L_DIM + q0 + ln;
    const _Float16* src = (hi == 0)
        ? &qc[(size_t)qrow * D_DIM + h * DH_DIM]
        : &qr[(size_t)qrow * DH_DIM];
#pragma unroll
    for (int j = 0; j < 8; ++j)
      *reinterpret_cast<v16h*>(&Qt[wave][ln * 256 + hi * 128 + j * 16]) =
          *reinterpret_cast<const v16h*>(&src[j * 16]);
  }

  v8f o[8];
#pragma unroll
  for (int nt = 0; nt < 8; ++nt) o[nt] = (v8f){};
  float mrow[8], srow[8];
#pragma unroll
  for (int j = 0; j < 8; ++j) { mrow[j] = -1e30f; srow[j] = 0.0f; }

  const size_t vrow0 = (size_t)(b * H_DIM + h) * DH_DIM;

  for (int kc0 = 0; kc0 < L_DIM; kc0 += 32) {
    const int key0 = b * L_DIM + kc0 + ln;
    const int key1 = key0 + 16;

    // ---- S = Q * K^T for two 16-key tiles (K-dim = 256 -> 8 WMMAs each)
    v8f s0 = {}, s1 = {};
#pragma unroll
    for (int kf = 0; kf < 8; ++kf) {
      const int kb = kf * 32 + hi * 16;  // lane's contiguous 16-half run
      const _Float16* k0p = (kb < DH_DIM)
          ? &kcm[(size_t)key0 * D_DIM + h * DH_DIM + kb]
          : &krm[(size_t)key0 * DH_DIM + (kb - DH_DIM)];
      const _Float16* k1p = (kb < DH_DIM)
          ? &kcm[(size_t)key1 * D_DIM + h * DH_DIM + kb]
          : &krm[(size_t)key1 * DH_DIM + (kb - DH_DIM)];
      v16h b0 = *reinterpret_cast<const v16h*>(k0p);
      v16h b1 = *reinterpret_cast<const v16h*>(k1p);

      v8h alo = *reinterpret_cast<const v8h*>(&Qt[wave][ln * 256 + kf * 32 + hi * 16]);
      v8h ahi = *reinterpret_cast<const v8h*>(&Qt[wave][ln * 256 + kf * 32 + hi * 16 + 8]);
      v16h qa = join8(alo, ahi);

      s0 = wmma_f16(qa, b0, s0);
      s1 = wmma_f16(qa, b1, s1);
    }

    // ---- scale + mask (lane's column = one key)
    const int mk0 = mask[b * L_DIM + kc0 + ln];
    const int mk1 = mask[b * L_DIM + kc0 + 16 + ln];
#pragma unroll
    for (int j = 0; j < 8; ++j) {
      s0[j] = mk0 ? s0[j] * ATTN_SCALE : -1e30f;
      s1[j] = mk1 ? s1[j] * ATTN_SCALE : -1e30f;
    }

    // ---- online softmax; row stats live across 16 lanes of a half-wave
    float alpha[8], p0[8], p1[8];
#pragma unroll
    for (int j = 0; j < 8; ++j) {
      float t = fmaxf(s0[j], s1[j]);
#pragma unroll
      for (int off = 1; off < 16; off <<= 1)
        t = fmaxf(t, __shfl_xor(t, off, 16));
      float mnew = fmaxf(mrow[j], t);
      alpha[j] = __expf(mrow[j] - mnew);
      mrow[j]  = mnew;
      p0[j] = __expf(s0[j] - mnew);
      p1[j] = __expf(s1[j] - mnew);
      float rs = p0[j] + p1[j];
#pragma unroll
      for (int off = 1; off < 16; off <<= 1)
        rs += __shfl_xor(rs, off, 16);
      srow[j] = srow[j] * alpha[j] + rs;
    }
#pragma unroll
    for (int nt = 0; nt < 8; ++nt)
#pragma unroll
      for (int j = 0; j < 8; ++j) o[nt][j] *= alpha[j];

    // ---- P: C-layout -> per-wave LDS -> A-layout fragment (in-order DS)
#pragma unroll
    for (int j = 0; j < 8; ++j) {
      int m = j + hi * 8;
      Pl[wave][m * 32 + ln]      = (_Float16)p0[j];
      Pl[wave][m * 32 + 16 + ln] = (_Float16)p1[j];
    }
    v8h plo = *reinterpret_cast<const v8h*>(&Pl[wave][ln * 32 + hi * 16]);
    v8h phi = *reinterpret_cast<const v8h*>(&Pl[wave][ln * 32 + hi * 16 + 8]);
    v16h pf = join8(plo, phi);

    // ---- O += P * V ; V fragment = single contiguous 32B load (V transposed)
#pragma unroll
    for (int nt = 0; nt < 8; ++nt) {
      v16h vf = *reinterpret_cast<const v16h*>(
          &vt[(vrow0 + nt * 16 + ln) * L_DIM + kc0 + hi * 16]);
      o[nt] = wmma_f16(pf, vf, o[nt]);
    }
  }

  // ---- normalize + write ctx
#pragma unroll
  for (int j = 0; j < 8; ++j) srow[j] = 1.0f / srow[j];
#pragma unroll
  for (int nt = 0; nt < 8; ++nt) {
#pragma unroll
    for (int j = 0; j < 8; ++j) {
      int qrow = b * L_DIM + q0 + j + hi * 8;
      ctx[(size_t)qrow * D_DIM + h * DH_DIM + nt * 16 + ln] =
          (_Float16)(o[nt][j] * srow[j]);
    }
  }
}

// ---------------------------------------------------------------------------
// Launcher
// ---------------------------------------------------------------------------
extern "C" void kernel_launch(void* const* d_in, const int* in_sizes, int n_in,
                              void* d_out, int out_size, void* d_ws, size_t ws_size,
                              hipStream_t stream) {
  (void)in_sizes; (void)n_in; (void)out_size; (void)ws_size;

  const float* x     = (const float*)d_in[0];
  const int*   amask = (const int*)d_in[1];
  const float* W_DKV = (const float*)d_in[2];
  const float* W_DQ  = (const float*)d_in[3];
  const float* W_UK  = (const float*)d_in[4];
  const float* W_UV  = (const float*)d_in[5];
  const float* W_UQ  = (const float*)d_in[6];
  const float* W_KR  = (const float*)d_in[7];
  const float* W_QR  = (const float*)d_in[8];
  const float* W_O   = (const float*)d_in[9];
  float* out = (float*)d_out;

  char* p = (char*)d_ws;
  auto alloc_h = [&](size_t elems) {
    _Float16* r = (_Float16*)p;
    p += ((elems * sizeof(_Float16) + 255) / 256) * 256;
    return r;
  };
  _Float16* xh     = alloc_h((size_t)ROWS * D_DIM);
  _Float16* wdkvt  = alloc_h((size_t)D_DIM * DC_DIM);   // [N=512,K=2048]
  _Float16* wdqt   = alloc_h((size_t)D_DIM * DC_DIM);
  _Float16* wukt   = alloc_h((size_t)DC_DIM * D_DIM);   // [N=2048,K=512]
  _Float16* wuvt   = alloc_h((size_t)DC_DIM * D_DIM);
  _Float16* wuqt   = alloc_h((size_t)DC_DIM * D_DIM);
  _Float16* wkrt   = alloc_h((size_t)D_DIM * DH_DIM);   // [128,2048]
  _Float16* wqrt   = alloc_h((size_t)DC_DIM * DH_DIM);  // [128,512]
  _Float16* wot    = alloc_h((size_t)D_DIM * D_DIM);    // [2048,2048]
  _Float16* ckv    = alloc_h((size_t)ROWS * DC_DIM);
  _Float16* cq     = alloc_h((size_t)ROWS * DC_DIM);
  _Float16* kcb    = alloc_h((size_t)ROWS * D_DIM);
  _Float16* vcb    = alloc_h((size_t)ROWS * D_DIM);
  _Float16* vtb    = alloc_h((size_t)ROWS * D_DIM);     // V transposed
  _Float16* qcb    = alloc_h((size_t)ROWS * D_DIM);
  _Float16* krpre  = alloc_h((size_t)ROWS * DH_DIM);
  _Float16* krb    = alloc_h((size_t)ROWS * DH_DIM);
  _Float16* qrpre  = alloc_h((size_t)ROWS * DH_DIM);
  _Float16* qrb    = alloc_h((size_t)ROWS * DH_DIM);
  _Float16* ctx    = alloc_h((size_t)ROWS * D_DIM);

  // convert x; transpose-convert all weights (LDS-tiled, fully coalesced)
  {
    size_t n = (size_t)ROWS * D_DIM;
    cvt_f32_to_f16<<<2048, 256, 0, stream>>>(x, xh, (int)n);
  }
  auto cvtT = [&](const float* src, _Float16* dst, int K, int N) {
    dim3 grid(K / 32, N / 32, 1);
    cvt_transpose_f16<<<grid, dim3(32, 8, 1), 0, stream>>>(src, dst, K, N);
  };
  cvtT(W_DKV, wdkvt, D_DIM, DC_DIM);
  cvtT(W_DQ,  wdqt,  D_DIM, DC_DIM);
  cvtT(W_UK,  wukt,  DC_DIM, D_DIM);
  cvtT(W_UV,  wuvt,  DC_DIM, D_DIM);
  cvtT(W_UQ,  wuqt,  DC_DIM, D_DIM);
  cvtT(W_KR,  wkrt,  D_DIM, DH_DIM);
  cvtT(W_QR,  wqrt,  DC_DIM, DH_DIM);
  cvtT(W_O,   wot,   D_DIM, D_DIM);

  auto gemm_h = [&](const _Float16* A, const _Float16* Bt, _Float16* C,
                    int M, int N, int K) {
    dim3 grid(N / 128, M / 64, 1);
    gemm_wmma_f16t<true><<<grid, 256, 0, stream>>>(A, Bt, (void*)C, M, N, K);
  };

  gemm_h(xh,  wdkvt, ckv,   ROWS, DC_DIM, D_DIM);   // c_kv = x @ W_DKV
  gemm_h(xh,  wdqt,  cq,    ROWS, DC_DIM, D_DIM);   // c_q  = x @ W_DQ
  gemm_h(ckv, wukt,  kcb,   ROWS, D_DIM,  DC_DIM);  // k_c  = c_kv @ W_UK
  gemm_h(ckv, wuvt,  vcb,   ROWS, D_DIM,  DC_DIM);  // v_c  = c_kv @ W_UV
  gemm_h(cq,  wuqt,  qcb,   ROWS, D_DIM,  DC_DIM);  // q_c  = c_q @ W_UQ
  gemm_h(xh,  wkrt,  krpre, ROWS, DH_DIM, D_DIM);   // x @ W_KR
  gemm_h(cq,  wqrt,  qrpre, ROWS, DH_DIM, DC_DIM);  // c_q @ W_QR

  // rope
  {
    int total = ROWS * 64;
    rope_apply<<<(total + 255) / 256, 256, 0, stream>>>(krpre, krb);
    rope_apply<<<(total + 255) / 256, 256, 0, stream>>>(qrpre, qrb);
  }

  // V transpose (key becomes fast axis), LDS-tiled
  {
    dim3 grid(L_DIM / 32, DH_DIM / 32, B_DIM * H_DIM);
    transpose_v_f16<<<grid, dim3(32, 8, 1), 0, stream>>>(vcb, vtb);
  }

  // fused attention
  {
    dim3 grid(L_DIM / 64, H_DIM, B_DIM);
    mla_flash_attn<<<grid, 128, 0, stream>>>(qcb, qrb, kcb, krb, vtb, amask, ctx);
  }

  // output projection (f32 straight to d_out)
  {
    dim3 grid(D_DIM / 128, ROWS / 64, 1);
    gemm_wmma_f16t<false><<<grid, 256, 0, stream>>>(ctx, wot, (void*)out,
                                                    ROWS, D_DIM, D_DIM);
  }
}
